// ParamReadout_18038862643635
// MI455X (gfx1250) — matrix operand order repaired and checked by reading, compiled, not run on hardware
//
#include <hip/hip_runtime.h>
#include <hip/hip_bf16.h>
#include <math.h>

typedef __bf16 bf16;
typedef __attribute__((ext_vector_type(16))) __bf16 v16bf;
typedef __attribute__((ext_vector_type(4)))  __bf16 bf16x4;
typedef __attribute__((ext_vector_type(8)))  float  v8f;

#define D_FEAT 128
#define TM 64          // entities per workgroup (2 waves x 32 rows)
#define NTHREADS 64    // 2 wave32, each wave owns TWO 16-row WMMA tiles

// ---------------------------------------------------------------------------
// WMMA helpers
// ---------------------------------------------------------------------------
__device__ __forceinline__ v8f wmma_bf16(v16bf a, v16bf b, v8f c) {
  // D = A(16x32 bf16) * B(32x16 bf16) + C(16x16 f32)
  return __builtin_amdgcn_wmma_f32_16x16x32_bf16(false, a, false, b,
                                                 (short)0, c, false, false);
}

// A-fragment element index inside a buffer laid out as
// [row_tile][k_chunk][lane(32)][16 bf16]  (one 32B v16bf per lane per chunk)
// 16-bit A 16x32 layout: lane = ((k>>3)&1)*16 + m ; elem within lane:
// e = ((k>>4)&1)*8 + ((k>>1)&3)*2 + (k&1)
__device__ __forceinline__ size_t afrag_addr(int row_tile, int nkc, int k, int m) {
  int lane = (((k >> 3) & 1) << 4) | (m & 15);
  int e    = (((k >> 4) & 1) << 3) | (((k >> 1) & 3) << 1) | (k & 1);
  return ((size_t)(row_tile * nkc + (k >> 5)) * 32 + lane) * 16 + e;
}

__device__ __forceinline__ v16bf load_afrag(const bf16* As, int row_tile, int kcn,
                                            int kc, int lane) {
  return *(const v16bf*)(As + ((size_t)(row_tile * kcn + kc) * 32 + lane) * 16);
}

// Two 16x16 output tiles sharing one B-fragment stream (2x B reuse).
// Optional block-reversed A (symmetrized pass: x_rev @ W == x @ P*W,
// P = 128-wide block reversal of K).
__device__ __forceinline__ void gemm_tile2(const bf16* __restrict__ Bw,
                                           const bf16* As, int t0, int t1,
                                           int lane, int n_tile, int kcn,
                                           bool rev, int arity,
                                           v8f& acc0, v8f& acc1) {
  const v16bf* bptr = (const v16bf*)Bw + (size_t)n_tile * kcn * 32 + lane;
  for (int kc = 0; kc < kcn; ++kc) {
    int akc = kc;
    if (rev) { int blk = kc >> 2; akc = (((arity - 1) - blk) << 2) | (kc & 3); }
    v16bf b = bptr[(size_t)kc * 32];
    if (kc + 1 < kcn)
      __builtin_prefetch((const void*)(bptr + (size_t)(kc + 1) * 32), 0, 1);
    v16bf a0 = load_afrag(As, t0, kcn, akc, lane);
    v16bf a1 = load_afrag(As, t1, kcn, akc, lane);
    acc0 = wmma_bf16(a0, b, acc0);
    acc1 = wmma_bf16(a1, b, acc1);
  }
}

// Broadcast per-column bias into the f32 C fragment (lane n = lane&15).
__device__ __forceinline__ v8f bias_acc(const float* __restrict__ b, int nc, int lane) {
  float bb = b[nc * 16 + (lane & 15)];
  v8f acc;
#pragma unroll
  for (int r = 0; r < 8; ++r) acc[r] = bb;
  return acc;
}

// Scatter a D fragment (16x16 f32) into A-fragment bf16 layout in LDS,
// with optional tanh activation. D layout: lane n = lane&15, m = (lane>>4)*8+r.
__device__ __forceinline__ void dfrag_to_afrag(bf16* dst, int row_tile, int lane,
                                               int nkc, int nc, v8f acc,
                                               bool do_tanh) {
  int n  = lane & 15;
  int mh = (lane >> 4) << 3;
#pragma unroll
  for (int r = 0; r < 8; ++r) {
    float v = acc[r];
    if (do_tanh) v = tanhf(v);
    dst[afrag_addr(row_tile, nkc, nc * 16 + n, mh + r)] = (bf16)v;
  }
}

// ---------------------------------------------------------------------------
// Weight prep: f32 (K x N) -> bf16 B-fragment layout [n_tile][k_chunk][lane][16]
// B 32x16 layout assumed: lane n = lane&15, k = (e>>3)*16 + (lane>>4)*8 + (e&7). Pads N->Npad.
// ---------------------------------------------------------------------------
__global__ void prep_weight_kernel(const float* __restrict__ W, bf16* __restrict__ dst,
                                   int K, int N, int Npad) {
  int kcn = K >> 5;
  int total = (Npad >> 4) * kcn * 512;
  for (int t = blockIdx.x * blockDim.x + threadIdx.x; t < total;
       t += gridDim.x * blockDim.x) {
    int e    = t & 15;
    int lane = (t >> 4) & 31;
    int rest = t >> 9;
    int kc   = rest % kcn;
    int nt   = rest / kcn;
    int k = (kc << 5) + ((e >> 3) << 4) + ((lane >> 4) << 3) + (e & 7);
    int n = (nt << 4) + (lane & 15);
    float v = (n < N) ? W[(size_t)k * N + n] : 0.0f;
    dst[t] = (bf16)v;
  }
}

// ---------------------------------------------------------------------------
// Misc small kernels
// ---------------------------------------------------------------------------
__global__ void zero_kernel(float* __restrict__ p, int n) {
  int i = blockIdx.x * blockDim.x + threadIdx.x;
  if (i < n) p[i] = 0.0f;
}

// Per (atom,d): segment-sum into h_mol + emit sigma/epsilon/q columns.
__global__ void atom_kernel(const float* __restrict__ h, const int* __restrict__ mol_id,
                            int na, float* __restrict__ hmol,
                            float* __restrict__ o_sig, float* __restrict__ o_eps,
                            float* __restrict__ o_q) {
  int t = blockIdx.x * blockDim.x + threadIdx.x;
  int atom = t >> 7, d = t & 127;
  if (atom >= na) return;
  float v = h[(size_t)atom * D_FEAT + d];
  atomicAdd(&hmol[(size_t)mol_id[atom] * D_FEAT + d], v);
  if (d == 0) o_sig[atom] = v;
  else if (d == 1) o_eps[atom] = v;
  else if (d == 2) o_q[atom] = v;
}

__global__ void pair_kernel(const float* __restrict__ h, const int* __restrict__ idx,
                            int n, float* __restrict__ o_eps, float* __restrict__ o_sig,
                            float* __restrict__ o_q) {
  int i = blockIdx.x * blockDim.x + threadIdx.x;
  if (i >= n) return;
  int a = idx[2 * i], b = idx[2 * i + 1];
  const float* ha = h + (size_t)a * D_FEAT;
  const float* hb = h + (size_t)b * D_FEAT;
  o_sig[i] = ha[0] * hb[0];
  o_eps[i] = sqrtf(fabsf(ha[1] * hb[1]));
  o_q[i]   = ha[2] * hb[2];
}

// ---------------------------------------------------------------------------
// Entity MLP:  out = MLP2( tanh-MLP(x) + tanh-MLP(x_rev) )
//   x = concat of `arity` gathered 128-wide rows of h  (Din = arity*128)
//   hidden: Din -> H (tanh) -> 128 (tanh), symmetrized over 2 passes
//   readout: 128 -> 512 (tanh) -> Nout (padded to 16 in prepped weights)
// Each wave owns two 16-row tiles so every B fragment feeds two WMMAs.
// ---------------------------------------------------------------------------
__global__ void __launch_bounds__(NTHREADS)
entity_mlp_kernel(const float* __restrict__ h, const int* __restrict__ idx,
                  int count, int arity,
                  const bf16* __restrict__ W1p, const float* __restrict__ b1, int H,
                  const bf16* __restrict__ W2p, const float* __restrict__ b2,
                  const bf16* __restrict__ Wr1p, const float* __restrict__ br1,
                  const bf16* __restrict__ Wr2p, const float* __restrict__ br2,
                  int Nout, float* __restrict__ outp) {
  extern __shared__ bf16 smem[];
  const int Din   = arity * D_FEAT;
  const int nkc_x = Din >> 5;
  const int nkc_y = H >> 5;
  bf16* xs = smem;                 // TM x Din  (A-fragment layout)
  bf16* ys = smem + TM * Din;      // TM x 512  (A-fragment layout, reused)
  const int lane = threadIdx.x & 31;
  const int wave = threadIdx.x >> 5;
  const int t0   = wave * 2;       // this wave's two row tiles
  const int t1   = wave * 2 + 1;
  const int row0 = blockIdx.x * TM;

  // ---- gather: f32 rows of h -> bf16 A fragments in LDS ----
  const int gpr   = Din >> 2;      // float4 groups per row
  const int total = TM * gpr;
  for (int t = threadIdx.x; t < total; t += blockDim.x) {
    int row = t / gpr;
    int k4  = (t - row * gpr) << 2;
    int rg = row0 + row; if (rg >= count) rg = count - 1;
    int atom = idx[(size_t)rg * arity + (k4 >> 7)];
    const float4 v = *(const float4*)(h + (size_t)atom * D_FEAT + (k4 & 127));
    bf16x4 p; p.x = (bf16)v.x; p.y = (bf16)v.y; p.z = (bf16)v.z; p.w = (bf16)v.w;
    *(bf16x4*)(xs + afrag_addr(row >> 4, nkc_x, k4, row & 15)) = p;
  }
  __syncthreads();

  v8f hsum0[8], hsum1[8];  // 32 x 128 symmetrized feature, f32 D fragments
#pragma unroll
  for (int i = 0; i < 8; ++i)
#pragma unroll
    for (int r = 0; r < 8; ++r) { hsum0[i][r] = 0.0f; hsum1[i][r] = 0.0f; }

  for (int pass = 0; pass < 2; ++pass) {
    const bool rev = (pass == 1);
    // GEMM1: x @ W1 + b1, tanh -> ys (A-frag)
    for (int nc = 0; nc < (H >> 4); ++nc) {
      v8f a0 = bias_acc(b1, nc, lane), a1 = a0;
      gemm_tile2(W1p, xs, t0, t1, lane, nc, nkc_x, rev, arity, a0, a1);
      dfrag_to_afrag(ys, t0, lane, nkc_y, nc, a0, true);
      dfrag_to_afrag(ys, t1, lane, nkc_y, nc, a1, true);
    }
    __syncthreads();
    // GEMM2: y @ W2 + b2, tanh, accumulate into hsum
    for (int nc = 0; nc < 8; ++nc) {
      v8f a0 = bias_acc(b2, nc, lane), a1 = a0;
      gemm_tile2(W2p, ys, t0, t1, lane, nc, nkc_y, false, 1, a0, a1);
#pragma unroll
      for (int r = 0; r < 8; ++r) {
        hsum0[nc][r] += tanhf(a0[r]);
        hsum1[nc][r] += tanhf(a1[r]);
      }
    }
    __syncthreads();
  }

  // ---- readout ----
  for (int nc = 0; nc < 8; ++nc) {        // hsum -> xs (A-frag, K=128)
    dfrag_to_afrag(xs, t0, lane, 4, nc, hsum0[nc], false);
    dfrag_to_afrag(xs, t1, lane, 4, nc, hsum1[nc], false);
  }
  __syncthreads();
  for (int nc = 0; nc < 32; ++nc) {       // GEMM3: 128 -> 512, tanh -> ys
    v8f a0 = bias_acc(br1, nc, lane), a1 = a0;
    gemm_tile2(Wr1p, xs, t0, t1, lane, nc, 4, false, 1, a0, a1);
    dfrag_to_afrag(ys, t0, lane, 16, nc, a0, true);
    dfrag_to_afrag(ys, t1, lane, 16, nc, a1, true);
  }
  __syncthreads();
  {                                       // GEMM4: 512 -> Nout (padded 16)
    int n = lane & 15;
    float bb = (n < Nout) ? br2[n] : 0.0f;
    v8f a0, a1;
#pragma unroll
    for (int r = 0; r < 8; ++r) { a0[r] = bb; a1[r] = bb; }
    gemm_tile2(Wr2p, ys, t0, t1, lane, 0, 16, false, 1, a0, a1);
    if (n < Nout) {
      int mh = (lane >> 4) << 3;
#pragma unroll
      for (int r = 0; r < 8; ++r) {
        int rg0 = row0 + (t0 << 4) + mh + r;
        int rg1 = row0 + (t1 << 4) + mh + r;
        if (rg0 < count) outp[(size_t)rg0 * Nout + n] = a0[r];
        if (rg1 < count) outp[(size_t)rg1 * Nout + n] = a1[r];
      }
    }
  }
}

// ---------------------------------------------------------------------------
// Molecule readout: rows(count x 128) -> tanh(rows@Wm1+b) @ Wm2 + b  (Nout=1)
// ---------------------------------------------------------------------------
__global__ void __launch_bounds__(NTHREADS)
mol_readout_kernel(const float* __restrict__ rows, int count,
                   const bf16* __restrict__ Wr1p, const float* __restrict__ br1,
                   const bf16* __restrict__ Wr2p, const float* __restrict__ br2,
                   int Nout, float* __restrict__ outp) {
  extern __shared__ bf16 smem[];
  bf16* xs = smem;                  // TM x 128
  bf16* ys = smem + TM * D_FEAT;    // TM x 512
  const int lane = threadIdx.x & 31;
  const int wave = threadIdx.x >> 5;
  const int t0   = wave * 2;
  const int t1   = wave * 2 + 1;
  const int row0 = blockIdx.x * TM;

  const int total = TM * (D_FEAT >> 2);
  for (int t = threadIdx.x; t < total; t += blockDim.x) {
    int row = t >> 5;
    int k4  = (t & 31) << 2;
    int rg = row0 + row; if (rg >= count) rg = count - 1;
    const float4 v = *(const float4*)(rows + (size_t)rg * D_FEAT + k4);
    bf16x4 p; p.x = (bf16)v.x; p.y = (bf16)v.y; p.z = (bf16)v.z; p.w = (bf16)v.w;
    *(bf16x4*)(xs + afrag_addr(row >> 4, 4, k4, row & 15)) = p;
  }
  __syncthreads();

  for (int nc = 0; nc < 32; ++nc) {
    v8f a0 = bias_acc(br1, nc, lane), a1 = a0;
    gemm_tile2(Wr1p, xs, t0, t1, lane, nc, 4, false, 1, a0, a1);
    dfrag_to_afrag(ys, t0, lane, 16, nc, a0, true);
    dfrag_to_afrag(ys, t1, lane, 16, nc, a1, true);
  }
  __syncthreads();
  {
    int n = lane & 15;
    float bb = (n < Nout) ? br2[n] : 0.0f;
    v8f a0, a1;
#pragma unroll
    for (int r = 0; r < 8; ++r) { a0[r] = bb; a1[r] = bb; }
    gemm_tile2(Wr2p, ys, t0, t1, lane, 0, 16, false, 1, a0, a1);
    if (n < Nout) {
      int mh = (lane >> 4) << 3;
#pragma unroll
      for (int r = 0; r < 8; ++r) {
        int rg0 = row0 + (t0 << 4) + mh + r;
        int rg1 = row0 + (t1 << 4) + mh + r;
        if (rg0 < count) outp[(size_t)rg0 * Nout + n] = a0[r];
        if (rg1 < count) outp[(size_t)rg1 * Nout + n] = a1[r];
      }
    }
  }
}

// ---------------------------------------------------------------------------
// Host side
// ---------------------------------------------------------------------------
extern "C" void kernel_launch(void* const* d_in, const int* in_sizes, int n_in,
                              void* d_out, int out_size, void* d_ws, size_t ws_size,
                              hipStream_t stream) {
  (void)n_in; (void)ws_size;
  const float* h    = (const float*)d_in[0];
  const float* Wb0a = (const float*)d_in[1];  const float* bb0a = (const float*)d_in[2];
  const float* Wb0b = (const float*)d_in[3];  const float* bb0b = (const float*)d_in[4];
  const float* Wa0a = (const float*)d_in[5];  const float* ba0a = (const float*)d_in[6];
  const float* Wa0b = (const float*)d_in[7];  const float* ba0b = (const float*)d_in[8];
  const float* Wt0a = (const float*)d_in[9];  const float* bt0a = (const float*)d_in[10];
  const float* Wt0b = (const float*)d_in[11]; const float* bt0b = (const float*)d_in[12];
  const float* Wb1  = (const float*)d_in[13]; const float* bb1  = (const float*)d_in[14];
  const float* Wb2  = (const float*)d_in[15]; const float* bb2  = (const float*)d_in[16];
  const float* Wa1  = (const float*)d_in[17]; const float* ba1  = (const float*)d_in[18];
  const float* Wa2  = (const float*)d_in[19]; const float* ba2  = (const float*)d_in[20];
  const float* Wt1  = (const float*)d_in[21]; const float* bt1  = (const float*)d_in[22];
  const float* Wt2  = (const float*)d_in[23]; const float* bt2  = (const float*)d_in[24];
  const float* Wm1  = (const float*)d_in[25]; const float* bm1  = (const float*)d_in[26];
  const float* Wm2  = (const float*)d_in[27]; const float* bm2  = (const float*)d_in[28];
  const int* bond_idx    = (const int*)d_in[29];
  const int* angle_idx   = (const int*)d_in[30];
  const int* torsion_idx = (const int*)d_in[31];
  const int* mol_id      = (const int*)d_in[32];
  const int* p14_idx     = (const int*)d_in[33];
  const int* nb_idx      = (const int*)d_in[34];

  const int NA   = in_sizes[0] / D_FEAT;
  const int NB   = in_sizes[29] / 2;
  const int NG   = in_sizes[30] / 3;
  const int NT   = in_sizes[31] / 4;
  const int NP14 = in_sizes[33] / 2;
  const int NNB  = in_sizes[34] / 2;
  const int NM   = out_size - (3 * NA + 4 * NB + 6 * NG + 14 * NT + 3 * NP14 + 3 * NNB);

  // workspace carve-up
  char* ws = (char*)d_ws;
  size_t off = 0;
  auto carve_bf = [&](int K, int Npad) -> bf16* {
    bf16* p = (bf16*)(ws + off);
    off += (((size_t)K * Npad * sizeof(bf16)) + 255) & ~(size_t)255;
    return p;
  };
  bf16* pWb0a = carve_bf(256, 256);
  bf16* pWb0b = carve_bf(256, 128);
  bf16* pWa0a = carve_bf(384, 384);
  bf16* pWa0b = carve_bf(384, 128);
  bf16* pWt0a = carve_bf(512, 512);
  bf16* pWt0b = carve_bf(512, 128);
  bf16* pWb1  = carve_bf(128, 512);
  bf16* pWb2  = carve_bf(512, 16);
  bf16* pWa1  = carve_bf(128, 512);
  bf16* pWa2  = carve_bf(512, 16);
  bf16* pWt1  = carve_bf(128, 512);
  bf16* pWt2  = carve_bf(512, 16);
  bf16* pWm1  = carve_bf(128, 512);
  bf16* pWm2  = carve_bf(512, 16);
  float* hmol = (float*)(ws + off);
  off += (size_t)NM * D_FEAT * sizeof(float);

  // output carve-up (reference tuple order, flattened)
  float* out    = (float*)d_out;
  float* o_sig  = out;
  float* o_eps  = out + (size_t)NA;
  float* o_q    = out + (size_t)2 * NA;
  float* o_bond = out + (size_t)3 * NA;
  float* o_ang  = o_bond + (size_t)NB * 4;
  float* o_tor  = o_ang  + (size_t)NG * 6;
  float* o_u0   = o_tor  + (size_t)NT * 14;
  float* o_e14  = o_u0  + NM;
  float* o_s14  = o_e14 + NP14;
  float* o_q14  = o_s14 + NP14;
  float* o_enb  = o_q14 + NP14;
  float* o_snb  = o_enb + NNB;
  float* o_qnb  = o_snb + NNB;

  // ---- 1) weight prep (f32 -> bf16 B-fragment layout, N padded to 16) ----
  struct WL { const float* src; bf16* dst; int K, N, Npad; };
  const WL wl[14] = {
      {Wb0a, pWb0a, 256, 256, 256}, {Wb0b, pWb0b, 256, 128, 128},
      {Wa0a, pWa0a, 384, 384, 384}, {Wa0b, pWa0b, 384, 128, 128},
      {Wt0a, pWt0a, 512, 512, 512}, {Wt0b, pWt0b, 512, 128, 128},
      {Wb1,  pWb1,  128, 512, 512}, {Wb2,  pWb2,  512,   4,  16},
      {Wa1,  pWa1,  128, 512, 512}, {Wa2,  pWa2,  512,   6,  16},
      {Wt1,  pWt1,  128, 512, 512}, {Wt2,  pWt2,  512,  14,  16},
      {Wm1,  pWm1,  128, 512, 512}, {Wm2,  pWm2,  512,   1,  16}};
  for (int i = 0; i < 14; ++i) {
    int total = (wl[i].Npad / 16) * (wl[i].K / 32) * 512;
    int blocks = (total + 255) / 256;
    prep_weight_kernel<<<blocks, 256, 0, stream>>>(wl[i].src, wl[i].dst,
                                                   wl[i].K, wl[i].N, wl[i].Npad);
  }

  // ---- 2) molecule segment sum + sigma/epsilon/q columns ----
  zero_kernel<<<(NM * D_FEAT + 255) / 256, 256, 0, stream>>>(hmol, NM * D_FEAT);
  atom_kernel<<<((size_t)NA * D_FEAT + 255) / 256, 256, 0, stream>>>(
      h, mol_id, NA, hmol, o_sig, o_eps, o_q);

  // ---- 3) pair products ----
  pair_kernel<<<(NP14 + 255) / 256, 256, 0, stream>>>(h, p14_idx, NP14, o_e14, o_s14, o_q14);
  pair_kernel<<<(NNB  + 255) / 256, 256, 0, stream>>>(h, nb_idx,  NNB,  o_enb, o_snb, o_qnb);

  // ---- 4) symmetrized entity MLPs (bond / angle / torsion) ----
  const dim3 blk(NTHREADS);
  entity_mlp_kernel<<<(NB + TM - 1) / TM, blk, (size_t)(TM * 256 + TM * 512) * sizeof(bf16), stream>>>(
      h, bond_idx, NB, 2, pWb0a, bb0a, 256, pWb0b, bb0b, pWb1, bb1, pWb2, bb2, 4, o_bond);
  entity_mlp_kernel<<<(NG + TM - 1) / TM, blk, (size_t)(TM * 384 + TM * 512) * sizeof(bf16), stream>>>(
      h, angle_idx, NG, 3, pWa0a, ba0a, 384, pWa0b, ba0b, pWa1, ba1, pWa2, ba2, 6, o_ang);
  entity_mlp_kernel<<<(NT + TM - 1) / TM, blk, (size_t)(TM * 512 + TM * 512) * sizeof(bf16), stream>>>(
      h, torsion_idx, NT, 4, pWt0a, bt0a, 512, pWt0b, bt0b, pWt1, bt1, pWt2, bt2, 14, o_tor);

  // ---- 5) molecule readout ----
  mol_readout_kernel<<<(NM + TM - 1) / TM, blk, (size_t)(TM * 128 + TM * 512) * sizeof(bf16), stream>>>(
      hmol, NM, pWm1, bm1, pWm2, bm2, 1, o_u0);
}
